// PatchVQVAETransformer_78099685310853
// MI455X (gfx1250) — compile-verified, matrix operand order checked by reading, not run on hardware
//
#include <hip/hip_runtime.h>
#include <cmath>

// ---------------- problem constants ----------------
#define BB        32
#define TT        8192
#define NROWS     16384            // BB * (TT/16)
#define CDIM      128
#define KCODES    1024
#define NLAYERS   4
#define DFFD      256
#define NHEADS    4
#define DHD       32
#define SEQ       512
#define LOGITS_N  (NROWS * KCODES)

typedef __attribute__((ext_vector_type(16))) __bf16 v16bf;
typedef __attribute__((ext_vector_type(8)))  float  v8f;

// ---------------- WMMA fragment helpers (wave32 layouts, ISA 7.12.2) --------
__device__ __forceinline__ v16bf frag_a_rm(const __bf16* A, int lda, int k0, int lane) {
  int m = lane & 15, half = lane >> 4;
  v16bf a;
#pragma unroll
  for (int i = 0; i < 16; ++i) {
    int k = ((i & 8) << 1) + half * 8 + (i & 7);
    a[i] = A[m * lda + k0 + k];
  }
  return a;
}
__device__ __forceinline__ v16bf frag_b_nk(const __bf16* Bp, int ldb, int k0, int lane) {
  int n = lane & 15, half = lane >> 4;
  v16bf b;
#pragma unroll
  for (int i = 0; i < 16; ++i) {
    int k = ((i & 8) << 1) + half * 8 + (i & 7);
    b[i] = Bp[n * ldb + k0 + k];
  }
  return b;
}
__device__ __forceinline__ v16bf frag_b_kn(const __bf16* Bp, int ld, int lane) {
  int n = lane & 15, half = lane >> 4;
  v16bf b;
#pragma unroll
  for (int i = 0; i < 16; ++i) {
    int k = ((i & 8) << 1) + half * 8 + (i & 7);
    b[i] = Bp[k * ld + n];
  }
  return b;
}
__device__ __forceinline__ v16bf frag_a_lds(const __bf16* pl, int lane) { // pl: 16x32
  int m = lane & 15, half = lane >> 4;
  v16bf a;
#pragma unroll
  for (int i = 0; i < 16; ++i) {
    int k = ((i & 8) << 1) + half * 8 + (i & 7);
    a[i] = pl[m * 32 + k];
  }
  return a;
}
__device__ __forceinline__ v8f wmma_bf16(v16bf a, v16bf b, v8f c) {
  return __builtin_amdgcn_wmma_f32_16x16x32_bf16(false, a, false, b, (short)0, c,
                                                 false, false);
}

// ------- async global->LDS staging (GLOBAL_LOAD_ASYNC_TO_LDS, ASYNCcnt) ----
// VDST VGPR carries the wave-relative LDS byte offset = low 32 bits of a flat
// pointer to a __shared__ object (ISA 10.2: LDS aperture uses addr[31:0]).
__device__ __forceinline__ void async_stage_b128(const void* gsrc, void* ldst) {
  unsigned l = (unsigned)(uintptr_t)ldst;
  asm volatile("global_load_async_to_lds_b128 %0, %1, off"
               :: "v"(l), "v"(gsrc) : "memory");
}
__device__ __forceinline__ void async_wait0() {
  asm volatile("s_wait_asynccnt 0x0" ::: "memory");
}

// ---------------- small utility kernels ----------------
__global__ void zero_loss_kernel(float* p) { *p = 0.f; }

__global__ void to_bf16_kernel(const float* __restrict__ src, __bf16* __restrict__ dst, int n) {
  int i = blockIdx.x * blockDim.x + threadIdx.x;
  if (i < n) dst[i] = (__bf16)src[i];
}

__global__ void cnorm_kernel(const float* __restrict__ cb, float* __restrict__ cn) {
  int i = blockIdx.x * blockDim.x + threadIdx.x;
  if (i < KCODES) {
    float s = 0.f;
    for (int d = 0; d < CDIM; ++d) { float v = cb[i * CDIM + d]; s += v * v; }
    cn[i] = s;
  }
}

// ---------------- encoder: per-patch conv stack (VALU + LDS) ----------------
__global__ __launch_bounds__(64)
void encoder_kernel(const float* __restrict__ x,
                    const float* __restrict__ w1, const float* __restrict__ b1,
                    const float* __restrict__ w2, const float* __restrict__ b2,
                    const float* __restrict__ w3, const float* __restrict__ b3,
                    const float* __restrict__ rw1, const float* __restrict__ rb1,
                    const float* __restrict__ rw2, const float* __restrict__ rb2,
                    const float* __restrict__ pw,  const float* __restrict__ pb,
                    float* __restrict__ z_f32, __bf16* __restrict__ z_bf,
                    float* __restrict__ znorm) {
  __shared__ float xs[16];
  __shared__ float bufA[64 * 8];
  __shared__ float bufB[64 * 8];
  __shared__ float zred[CDIM];
  int n = blockIdx.x;
  int tid = threadIdx.x;
  int b = n >> 9, p = n & 511;
  if (tid < 16) xs[tid] = x[(size_t)b * TT + p * 16 + tid];
  __syncthreads();
  if (tid < 32) {                       // conv1 1->32, k4 s2 p1, relu
    for (int j = 0; j < 8; ++j) {
      float acc = b1[tid];
      for (int kk = 0; kk < 4; ++kk) {
        int t = 2 * j + kk - 1;
        if (t >= 0 && t < 16) acc += w1[tid * 4 + kk] * xs[t];
      }
      bufA[tid * 8 + j] = fmaxf(acc, 0.f);
    }
  }
  __syncthreads();
  for (int j = 0; j < 4; ++j) {         // conv2 32->64, k4 s2 p1, relu
    float acc = b2[tid];
    for (int ci = 0; ci < 32; ++ci)
      for (int kk = 0; kk < 4; ++kk) {
        int t = 2 * j + kk - 1;
        if (t >= 0 && t < 8) acc += w2[(tid * 32 + ci) * 4 + kk] * bufA[ci * 8 + t];
      }
    bufB[tid * 8 + j] = fmaxf(acc, 0.f);
  }
  __syncthreads();
  for (int j = 0; j < 4; ++j) {         // conv3 64->64, k3 s1 p1
    float acc = b3[tid];
    for (int ci = 0; ci < 64; ++ci)
      for (int kk = 0; kk < 3; ++kk) {
        int t = j + kk - 1;
        if (t >= 0 && t < 4) acc += w3[(tid * 64 + ci) * 3 + kk] * bufB[ci * 8 + t];
      }
    bufA[tid * 8 + j] = acc;
  }
  __syncthreads();
  for (int L = 0; L < 2; ++L) {         // 2 residual layers
    if (tid < 32) {
      for (int j = 0; j < 4; ++j) {
        float acc = rb1[L * 32 + tid];
        for (int ci = 0; ci < 64; ++ci)
          for (int kk = 0; kk < 3; ++kk) {
            int t = j + kk - 1;
            if (t >= 0 && t < 4)
              acc += rw1[((L * 32 + tid) * 64 + ci) * 3 + kk] * fmaxf(bufA[ci * 8 + t], 0.f);
          }
        bufB[tid * 8 + j] = fmaxf(acc, 0.f);
      }
    }
    __syncthreads();
    for (int j = 0; j < 4; ++j) {
      float acc = rb2[L * 64 + tid];
      for (int ci = 0; ci < 32; ++ci)
        acc += rw2[(L * 64 + tid) * 32 + ci] * bufB[ci * 8 + j];
      bufA[tid * 8 + j] += acc;
    }
    __syncthreads();
  }
  if (tid < 32) {                       // pre conv1x1 over relu(h): 64 -> 32
    for (int j = 0; j < 4; ++j) {
      float acc = pb[tid];
      for (int ci = 0; ci < 64; ++ci)
        acc += pw[tid * 64 + ci] * fmaxf(bufA[ci * 8 + j], 0.f);
      int d = tid * 4 + j;
      z_f32[(size_t)n * CDIM + d] = acc;
      z_bf[(size_t)n * CDIM + d] = (__bf16)acc;
      zred[d] = acc * acc;
    }
  }
  __syncthreads();
  if (tid == 0) {
    float s = 0.f;
    for (int d = 0; d < CDIM; ++d) s += zred[d];
    znorm[n] = s;
  }
}

// ------- VQ: z resident in regs, codebook async-staged in LDS --------------
__global__ __launch_bounds__(256)
void vq_kernel(const __bf16* __restrict__ z_bf, const float* __restrict__ z_f32,
               const float* __restrict__ znorm, const __bf16* __restrict__ cb_bf,
               const float* __restrict__ cnorm, const float* __restrict__ codebook,
               const float* __restrict__ pos_emb,
               float* __restrict__ hh_f32, __bf16* __restrict__ hh_bf,
               float* __restrict__ out_loss, float* __restrict__ out_idx) {
  __shared__ __align__(16) __bf16 cpan[64 * CDIM];   // 16 KB codebook panel
  __shared__ int idx_s[8][16];
  int tid = threadIdx.x;
  int lane = tid & 31, wv = tid >> 5;
  int mt = blockIdx.x * 8 + wv;                      // 16-row tile per wave
  int half = lane >> 4, nl = lane & 15;

  v16bf az[CDIM / 32];
#pragma unroll
  for (int kc = 0; kc < CDIM / 32; ++kc)
    az[kc] = frag_a_rm(z_bf + (size_t)mt * 16 * CDIM, CDIM, kc * 32, lane);

  float zn[8];
#pragma unroll
  for (int j = 0; j < 8; ++j) zn[j] = znorm[mt * 16 + j + 8 * half];

  float minv[8]; int mini[8];
#pragma unroll
  for (int j = 0; j < 8; ++j) { minv[j] = 3.4e38f; mini[j] = 0; }

  for (int panel = 0; panel < KCODES / 64; ++panel) {
    __syncthreads();   // previous panel fully consumed
    {
      const char* src = (const char*)(cb_bf + (size_t)panel * 64 * CDIM);
      for (int i = tid; i < 64 * CDIM / 8; i += 256)
        async_stage_b128(src + (size_t)i * 16, (char*)cpan + (size_t)i * 16);
      async_wait0();
    }
    __syncthreads();
    v8f acc[4] = {};
#pragma unroll
    for (int kc = 0; kc < CDIM / 32; ++kc) {
      v16bf bf[4];
#pragma unroll
      for (int sub = 0; sub < 4; ++sub)
        bf[sub] = frag_b_nk(cpan + sub * 16 * CDIM, CDIM, kc * 32, lane);
#pragma unroll
      for (int sub = 0; sub < 4; ++sub)
        acc[sub] = wmma_bf16(az[kc], bf[sub], acc[sub]);  // 4 independent chains
    }
#pragma unroll
    for (int sub = 0; sub < 4; ++sub) {
      int nc = panel * 64 + sub * 16 + nl;
      float cn = cnorm[nc];
#pragma unroll
      for (int j = 0; j < 8; ++j) {
        float d = zn[j] + cn - 2.f * acc[sub][j];
        if (d < minv[j]) { minv[j] = d; mini[j] = nc; }
      }
    }
  }
  for (int mask = 1; mask <= 8; mask <<= 1) {
#pragma unroll
    for (int j = 0; j < 8; ++j) {
      float ov = __shfl_xor(minv[j], mask, 32);
      int   oi = __shfl_xor(mini[j], mask, 32);
      if (ov < minv[j] || (ov == minv[j] && oi < mini[j])) { minv[j] = ov; mini[j] = oi; }
    }
  }
  if (nl == 0)
#pragma unroll
    for (int j = 0; j < 8; ++j) idx_s[wv][j + 8 * half] = mini[j];
  __syncthreads();
  if (lane < 16) {
    int m = mt * 16 + lane;
    int id = idx_s[wv][lane];
    int s = m & (SEQ - 1);
    float lsum = 0.f;
    for (int d = 0; d < CDIM; ++d) {
      float q = codebook[id * CDIM + d];
      float zv = z_f32[(size_t)m * CDIM + d];
      float dd = q - zv; lsum += dd * dd;
      float hv = q + pos_emb[s * CDIM + d];
      hh_f32[(size_t)m * CDIM + d] = hv;
      hh_bf[(size_t)m * CDIM + d] = (__bf16)hv;
    }
    atomicAdd(out_loss, lsum * (1.25f / ((float)NROWS * (float)CDIM)));
    out_idx[m] = (float)id;
  }
}

// ---- GEMM: Y = act(X * W^T + b); 16x64 per wave, W panel async in LDS -----
__device__ __forceinline__ float gelu_exact(float v) {
  return 0.5f * v * (1.f + erff(v * 0.70710678118654752f));
}

template <int ACT, int KD>  // ACT: 0 none, 1 exact GELU
__global__ __launch_bounds__(256)
void gemm_bias_kernel(const __bf16* __restrict__ A, const __bf16* __restrict__ W,
                      const float* __restrict__ bias,
                      float* __restrict__ Cf, __bf16* __restrict__ Cb,
                      int M, int N) {
  __shared__ __align__(16) __bf16 wpanel[64 * KD];   // 64-row weight panel
  int tid = threadIdx.x;
  int lane = tid & 31, wv = tid >> 5;
  int npanels = N >> 6;
  int msuper = blockIdx.x / npanels;                  // 128 rows
  int np     = blockIdx.x % npanels;                  // 64 cols

  // async-stage the weight panel into LDS (no VGPR round-trip)
  {
    const char* src = (const char*)(W + (size_t)np * 64 * KD);
    for (int i = tid; i < 64 * KD / 8; i += 256)
      async_stage_b128(src + (size_t)i * 16, (char*)wpanel + (size_t)i * 16);
  }

  // A fragments: this wave's 16 rows, full K, held in registers (overlaps
  // with the in-flight async panel loads)
  const __bf16* Ap = A + (size_t)(msuper * 128 + wv * 16) * KD;
  v16bf afr[KD / 32];
#pragma unroll
  for (int kc = 0; kc < KD / 32; ++kc)
    afr[kc] = frag_a_rm(Ap, KD, kc * 32, lane);

  // prefetch next weight panel (global_prefetch_b8)
  if (np + 1 < npanels) {
    const char* nw = (const char*)(W + (size_t)(np + 1) * 64 * KD);
    for (int i = tid; i < (64 * KD * 2) / 256; i += 256)
      __builtin_prefetch(nw + (size_t)i * 256, 0, 1);
  }
  async_wait0();
  __syncthreads();

  v8f acc[4] = {};
#pragma unroll
  for (int kc = 0; kc < KD / 32; ++kc) {
    v16bf bf[4];
#pragma unroll
    for (int sub = 0; sub < 4; ++sub)
      bf[sub] = frag_b_nk(wpanel + sub * 16 * KD, KD, kc * 32, lane);
#pragma unroll
    for (int sub = 0; sub < 4; ++sub)
      acc[sub] = wmma_bf16(afr[kc], bf[sub], acc[sub]);  // 4 independent chains
  }

  int half = lane >> 4;
#pragma unroll
  for (int sub = 0; sub < 4; ++sub) {
    int n = (np << 6) + (sub << 4) + (lane & 15);
    float bv = bias ? bias[n] : 0.f;
#pragma unroll
    for (int j = 0; j < 8; ++j) {
      int m = msuper * 128 + wv * 16 + j + 8 * half;
      float v = acc[sub][j] + bv;
      if (ACT == 1) v = gelu_exact(v);
      if (Cf) Cf[(size_t)m * N + n] = v;
      if (Cb) Cb[(size_t)m * N + n] = (__bf16)v;
    }
  }
}

// ---------------- flash-style causal attention (one wave / q-tile) --------
__global__ __launch_bounds__(32)
void attn_kernel(const __bf16* __restrict__ qkv, __bf16* __restrict__ attn_out) {
  __shared__ __align__(16) __bf16 pl[16 * 32];
  int lane = threadIdx.x;
  int bid = blockIdx.x;                  // B*H*(SEQ/16) = 4096
  int qt = bid & 31;
  int h  = (bid >> 5) & 3;
  int b  = bid >> 7;
  const float scale = 0.1767766952966369f;  // 1/sqrt(32)
  int row0 = b * SEQ + qt * 16;
  int half = lane >> 4, nl = lane & 15;

  v16bf aq = frag_a_rm(qkv + (size_t)row0 * 384 + h * DHD, 384, 0, lane);
  v8f o0 = {}, o1 = {};
  float rmax[8], rsum[8];
#pragma unroll
  for (int j = 0; j < 8; ++j) { rmax[j] = -1e30f; rsum[j] = 0.f; }

  int qend = qt * 16 + 15;
  for (int kk0 = 0; kk0 <= qend; kk0 += 32) {
    float p[2][8], tmax[8];
#pragma unroll
    for (int j = 0; j < 8; ++j) tmax[j] = -1e30f;
    for (int sub = 0; sub < 2; ++sub) {
      int n0 = kk0 + sub * 16;
      v8f sf = {};
      if (n0 <= qend) {
        v16bf bk = frag_b_nk(qkv + (size_t)(b * SEQ + n0) * 384 + CDIM + h * DHD,
                             384, 0, lane);
        sf = wmma_bf16(aq, bk, sf);
      }
#pragma unroll
      for (int j = 0; j < 8; ++j) {
        int qi = qt * 16 + j + 8 * half;
        int kj = n0 + nl;
        float v = (n0 <= qend && kj <= qi) ? sf[j] * scale : -1e30f;
        p[sub][j] = v;
        tmax[j] = fmaxf(tmax[j], v);
      }
    }
    for (int mask = 1; mask <= 8; mask <<= 1)
#pragma unroll
      for (int j = 0; j < 8; ++j)
        tmax[j] = fmaxf(tmax[j], __shfl_xor(tmax[j], mask, 32));
    float tsum[8];
#pragma unroll
    for (int j = 0; j < 8; ++j) {
      float nm = fmaxf(rmax[j], tmax[j]);
      float corr = __expf(rmax[j] - nm);
      float e0 = __expf(p[0][j] - nm);
      float e1 = __expf(p[1][j] - nm);
      p[0][j] = e0; p[1][j] = e1;
      tsum[j] = e0 + e1;
      rsum[j] *= corr;
      o0[j] *= corr; o1[j] *= corr;
      rmax[j] = nm;
    }
    for (int mask = 1; mask <= 8; mask <<= 1)
#pragma unroll
      for (int j = 0; j < 8; ++j) tsum[j] += __shfl_xor(tsum[j], mask, 32);
#pragma unroll
    for (int j = 0; j < 8; ++j) rsum[j] += tsum[j];
#pragma unroll
    for (int sub = 0; sub < 2; ++sub)
#pragma unroll
      for (int j = 0; j < 8; ++j)
        pl[(j + 8 * half) * 32 + sub * 16 + nl] = (__bf16)p[sub][j];
    __syncthreads();
    v16bf ap = frag_a_lds(pl, lane);
    const __bf16* Vp = qkv + (size_t)(b * SEQ + kk0) * 384 + 2 * CDIM + h * DHD;
    v16bf bv0 = frag_b_kn(Vp, 384, lane);
    v16bf bv1 = frag_b_kn(Vp + 16, 384, lane);
    o0 = wmma_bf16(ap, bv0, o0);
    o1 = wmma_bf16(ap, bv1, o1);
    __syncthreads();
  }
#pragma unroll
  for (int j = 0; j < 8; ++j) {
    float inv = 1.f / rsum[j];
    int m = row0 + j + 8 * half;
    attn_out[(size_t)m * CDIM + h * DHD + nl]      = (__bf16)(o0[j] * inv);
    attn_out[(size_t)m * CDIM + h * DHD + 16 + nl] = (__bf16)(o1[j] * inv);
  }
}

// ---------------- residual add + LayerNorm (one wave per row) --------------
template <bool RES>
__global__ __launch_bounds__(256)
void addln_kernel(const float* __restrict__ x, const float* __restrict__ r,
                  const float* __restrict__ g, const float* __restrict__ bb,
                  float* __restrict__ yf, __bf16* __restrict__ yb) {
  int row = blockIdx.x * (blockDim.x >> 5) + (threadIdx.x >> 5);
  int lane = threadIdx.x & 31;
  float v[4];
#pragma unroll
  for (int i = 0; i < 4; ++i) {
    int c = lane + 32 * i;
    v[i] = x[(size_t)row * CDIM + c] + (RES ? r[(size_t)row * CDIM + c] : 0.f);
  }
  float s = v[0] + v[1] + v[2] + v[3];
  for (int m = 1; m < 32; m <<= 1) s += __shfl_xor(s, m, 32);
  float mean = s * (1.f / 128.f);
  float q = 0.f;
#pragma unroll
  for (int i = 0; i < 4; ++i) { float d = v[i] - mean; q += d * d; }
  for (int m = 1; m < 32; m <<= 1) q += __shfl_xor(q, m, 32);
  float rstd = rsqrtf(q * (1.f / 128.f) + 1e-5f);
#pragma unroll
  for (int i = 0; i < 4; ++i) {
    int c = lane + 32 * i;
    float o = (v[i] - mean) * rstd * g[c] + bb[c];
    if (yf) yf[(size_t)row * CDIM + c] = o;
    if (yb) yb[(size_t)row * CDIM + c] = (__bf16)o;
  }
}

// ---------------- launch ----------------
static inline int gemm_blocks2(int M, int N) { return (M / 128) * (N / 64); }

extern "C" void kernel_launch(void* const* d_in, const int* in_sizes, int n_in,
                              void* d_out, int out_size, void* d_ws, size_t ws_size,
                              hipStream_t stream) {
  const float* x      = (const float*)d_in[0];
  const float* enc_w1 = (const float*)d_in[1];
  const float* enc_b1 = (const float*)d_in[2];
  const float* enc_w2 = (const float*)d_in[3];
  const float* enc_b2 = (const float*)d_in[4];
  const float* enc_w3 = (const float*)d_in[5];
  const float* enc_b3 = (const float*)d_in[6];
  const float* res_w1 = (const float*)d_in[7];
  const float* res_b1 = (const float*)d_in[8];
  const float* res_w2 = (const float*)d_in[9];
  const float* res_b2 = (const float*)d_in[10];
  const float* pre_w  = (const float*)d_in[11];
  const float* pre_b  = (const float*)d_in[12];
  const float* codebook = (const float*)d_in[13];
  const float* pos_emb  = (const float*)d_in[14];
  const float* qkv_w  = (const float*)d_in[15];
  const float* qkv_b  = (const float*)d_in[16];
  const float* proj_w = (const float*)d_in[17];
  const float* proj_b = (const float*)d_in[18];
  const float* ln1_g  = (const float*)d_in[19];
  const float* ln1_b  = (const float*)d_in[20];
  const float* ln2_g  = (const float*)d_in[21];
  const float* ln2_b  = (const float*)d_in[22];
  const float* ffn_w1 = (const float*)d_in[23];
  const float* ffn_b1 = (const float*)d_in[24];
  const float* ffn_w2 = (const float*)d_in[25];
  const float* ffn_b2 = (const float*)d_in[26];
  const float* fin_g  = (const float*)d_in[27];
  const float* fin_b  = (const float*)d_in[28];
  const float* head_w = (const float*)d_in[29];
  const float* head_b = (const float*)d_in[30];

  float* out = (float*)d_out;
  float* logits = out;
  float* loss_p = out + LOGITS_N;
  float* idx_p  = out + LOGITS_N + 1;

  char* ws = (char*)d_ws;
  size_t off = 0;
  auto alloc = [&](size_t bytes) -> void* {
    void* p = ws + off;
    off = (off + bytes + 255) & ~(size_t)255;
    return p;
  };
  float*  z_f32   = (float*) alloc((size_t)NROWS * CDIM * 4);
  __bf16* z_bf    = (__bf16*)alloc((size_t)NROWS * CDIM * 2);
  float*  znorm   = (float*) alloc((size_t)NROWS * 4);
  float*  cnorm   = (float*) alloc((size_t)KCODES * 4);
  __bf16* cb_bf   = (__bf16*)alloc((size_t)KCODES * CDIM * 2);
  __bf16* qkvw_bf = (__bf16*)alloc((size_t)NLAYERS * 3 * CDIM * CDIM * 2);
  __bf16* projw_bf= (__bf16*)alloc((size_t)NLAYERS * CDIM * CDIM * 2);
  __bf16* f1w_bf  = (__bf16*)alloc((size_t)NLAYERS * DFFD * CDIM * 2);
  __bf16* f2w_bf  = (__bf16*)alloc((size_t)NLAYERS * CDIM * DFFD * 2);
  __bf16* headw_bf= (__bf16*)alloc((size_t)KCODES * CDIM * 2);
  float*  hh_f32  = (float*) alloc((size_t)NROWS * CDIM * 4);
  __bf16* hh_bf   = (__bf16*)alloc((size_t)NROWS * CDIM * 2);
  __bf16* qkv_bf  = (__bf16*)alloc((size_t)NROWS * 3 * CDIM * 2);
  __bf16* attn_bf = (__bf16*)alloc((size_t)NROWS * CDIM * 2);
  float*  tmp_f32 = (float*) alloc((size_t)NROWS * CDIM * 4);
  __bf16* f1_bf   = (__bf16*)alloc((size_t)NROWS * DFFD * 2);
  __bf16* fin_bf  = (__bf16*)alloc((size_t)NROWS * CDIM * 2);
  (void)ws_size; (void)in_sizes; (void)n_in; (void)out_size;

  zero_loss_kernel<<<1, 1, 0, stream>>>(loss_p);
  auto conv = [&](const float* s, __bf16* d, int n) {
    to_bf16_kernel<<<(n + 255) / 256, 256, 0, stream>>>(s, d, n);
  };
  conv(codebook, cb_bf, KCODES * CDIM);
  conv(qkv_w, qkvw_bf, NLAYERS * 3 * CDIM * CDIM);
  conv(proj_w, projw_bf, NLAYERS * CDIM * CDIM);
  conv(ffn_w1, f1w_bf, NLAYERS * DFFD * CDIM);
  conv(ffn_w2, f2w_bf, NLAYERS * CDIM * DFFD);
  conv(head_w, headw_bf, KCODES * CDIM);
  cnorm_kernel<<<(KCODES + 255) / 256, 256, 0, stream>>>(codebook, cnorm);

  encoder_kernel<<<NROWS, 64, 0, stream>>>(
      x, enc_w1, enc_b1, enc_w2, enc_b2, enc_w3, enc_b3,
      res_w1, res_b1, res_w2, res_b2, pre_w, pre_b, z_f32, z_bf, znorm);

  vq_kernel<<<NROWS / 128, 256, 0, stream>>>(
      z_bf, z_f32, znorm, cb_bf, cnorm, codebook, pos_emb,
      hh_f32, hh_bf, loss_p, idx_p);

  for (int i = 0; i < NLAYERS; ++i) {
    gemm_bias_kernel<0, CDIM><<<gemm_blocks2(NROWS, 3 * CDIM), 256, 0, stream>>>(
        hh_bf, qkvw_bf + (size_t)i * 3 * CDIM * CDIM, qkv_b + i * 3 * CDIM,
        nullptr, qkv_bf, NROWS, 3 * CDIM);
    attn_kernel<<<BB * NHEADS * (SEQ / 16), 32, 0, stream>>>(qkv_bf, attn_bf);
    gemm_bias_kernel<0, CDIM><<<gemm_blocks2(NROWS, CDIM), 256, 0, stream>>>(
        attn_bf, projw_bf + (size_t)i * CDIM * CDIM, proj_b + i * CDIM,
        tmp_f32, nullptr, NROWS, CDIM);
    addln_kernel<true><<<NROWS / 8, 256, 0, stream>>>(
        hh_f32, tmp_f32, ln1_g + i * CDIM, ln1_b + i * CDIM, hh_f32, hh_bf);
    gemm_bias_kernel<1, CDIM><<<gemm_blocks2(NROWS, DFFD), 256, 0, stream>>>(
        hh_bf, f1w_bf + (size_t)i * DFFD * CDIM, ffn_b1 + i * DFFD,
        nullptr, f1_bf, NROWS, DFFD);
    gemm_bias_kernel<0, DFFD><<<gemm_blocks2(NROWS, CDIM), 256, 0, stream>>>(
        f1_bf, f2w_bf + (size_t)i * CDIM * DFFD, ffn_b2 + i * CDIM,
        tmp_f32, nullptr, NROWS, CDIM);
    addln_kernel<true><<<NROWS / 8, 256, 0, stream>>>(
        hh_f32, tmp_f32, ln2_g + i * CDIM, ln2_b + i * CDIM, hh_f32, hh_bf);
  }

  addln_kernel<false><<<NROWS / 8, 256, 0, stream>>>(
      hh_f32, nullptr, fin_g, fin_b, nullptr, fin_bf);
  gemm_bias_kernel<0, CDIM><<<gemm_blocks2(NROWS, KCODES), 256, 0, stream>>>(
      fin_bf, headw_bf, head_b, logits, nullptr, NROWS, KCODES);
}
